// MultiheadLatentAttention_13262859010620
// MI455X (gfx1250) — matrix operand order repaired and checked by reading, not compiled, not run on hardware
//
#include <hip/hip_runtime.h>

// ---------------------------------------------------------------------------
// MLA forward for gfx1250 (MI455X): all GEMM-shaped work on v_wmma_f32_16x16x32_f16
// ---------------------------------------------------------------------------

typedef __attribute__((ext_vector_type(16))) _Float16 v16h;
typedef __attribute__((ext_vector_type(8)))  _Float16 h8;
typedef __attribute__((ext_vector_type(4)))  _Float16 h4;
typedef __attribute__((ext_vector_type(8)))  float    v8f;
typedef __attribute__((ext_vector_type(4)))  float    f4;

using half_t = _Float16;

__device__ __forceinline__ v16h pack16(h8 lo, h8 hi) {
  v16h r;
#pragma unroll
  for (int i = 0; i < 8; ++i) { r[i] = lo[i]; r[i + 8] = hi[i]; }
  return r;
}

// ---------------------------------------------------------------------------
// Tiled WMMA GEMM: C[M,N] = A[M,K] * B[K,N]
//   block tile 64x128, 8 waves, each wave 32x32 (2x2 WMMA accum), BK=32.
//   A: f32 or f16 (converted to f16 into LDS), B: f32 weights (-> f16 LDS,
//   stored transposed [n][k] so B-fragments are contiguous ds_load_b128).
//   M must be multiple of 64, K multiple of 32; N is guard-checked.
// ---------------------------------------------------------------------------
template <typename AT, bool OUT_HALF>
__global__ __launch_bounds__(256) void gemm_wmma(
    const AT* __restrict__ A, const float* __restrict__ Bw,
    void* __restrict__ Cv, int M, int N, int K) {
  __shared__ alignas(16) half_t As[64 * 32];
  __shared__ alignas(16) half_t Bs[128 * 32];

  const int tid  = threadIdx.x;
  const int lane = tid & 31;
  const int wave = tid >> 5;
  const int wm   = (wave >> 2) * 32;   // 0 or 32
  const int wn   = (wave & 3) * 32;    // 0..96
  const int m0   = blockIdx.y * 64;
  const int n0   = blockIdx.x * 128;
  const int hl   = lane >> 4;          // half-wave id
  const int l16  = lane & 15;

  v8f acc[2][2] = {};

  const int arow = tid >> 2;           // 0..63
  const int acol = (tid & 3) * 8;      // 0,8,16,24
  const int bk   = tid >> 3;           // 0..31  (k row of B tile)
  const int bn   = (tid & 7) * 16;     // 0..112 (n chunk of B tile)

  for (int k0 = 0; k0 < K; k0 += 32) {
    // ---- stage A tile (64x32) to LDS, converting to f16 ----
    {
      const AT* src = A + (size_t)(m0 + arow) * K + (k0 + acol);
      h8 hv;
      if constexpr (sizeof(AT) == 4) {
        const f4* s4 = (const f4*)src;
        f4 x0 = s4[0], x1 = s4[1];
#pragma unroll
        for (int i = 0; i < 4; ++i) {
          hv[i]     = (half_t)x0[i];
          hv[i + 4] = (half_t)x1[i];
        }
      } else {
        hv = *(const h8*)src;
      }
      *(h8*)&As[arow * 32 + acol] = hv;
      if (k0 + 32 < K) __builtin_prefetch(src + 32, 0, 1);
    }
    // ---- stage B tile (32x128) to LDS transposed [n][k] ----
    {
      const float* srow = Bw + (size_t)(k0 + bk) * N;
      if (n0 + bn + 16 <= N) {
        const f4* s4 = (const f4*)(srow + n0 + bn);
#pragma unroll
        for (int q = 0; q < 4; ++q) {
          f4 xv = s4[q];
#pragma unroll
          for (int j = 0; j < 4; ++j)
            Bs[(bn + q * 4 + j) * 32 + bk] = (half_t)xv[j];
        }
      } else {
#pragma unroll
        for (int j = 0; j < 16; ++j) {
          int n = n0 + bn + j;
          Bs[(bn + j) * 32 + bk] = (n < N) ? (half_t)srow[n] : (half_t)0.0f;
        }
      }
    }
    __syncthreads();

    // ---- build fragments and issue WMMAs ----
    v16h af[2], bf[2];
#pragma unroll
    for (int i = 0; i < 2; ++i) {
      // A fragment: row = lane%16; K chunks at 8*hl and 16+8*hl (ISA layout)
      const h8* p = (const h8*)&As[(wm + 16 * i + l16) * 32];
      af[i] = pack16(p[hl], p[2 + hl]);
    }
#pragma unroll
    for (int j = 0; j < 2; ++j) {
      // B fragment: N = lane%16; K chunks at 8*hl and 16+8*hl (ISA layout)
      const h8* p = (const h8*)&Bs[(wn + 16 * j + l16) * 32];
      bf[j] = pack16(p[hl], p[2 + hl]);
    }
#pragma unroll
    for (int i = 0; i < 2; ++i)
#pragma unroll
      for (int j = 0; j < 2; ++j)
        acc[i][j] = __builtin_amdgcn_wmma_f32_16x16x32_f16(
            false, af[i], false, bf[j], (short)0, acc[i][j], false, false);
    __syncthreads();
  }

  // ---- writeback (C layout: M = r + 8*hl, N = lane%16) ----
#pragma unroll
  for (int i = 0; i < 2; ++i)
#pragma unroll
    for (int j = 0; j < 2; ++j)
#pragma unroll
      for (int r = 0; r < 8; ++r) {
        int rg = m0 + wm + 16 * i + r + 8 * hl;
        int cg = n0 + wn + 16 * j + l16;
        if (cg < N) {
          if constexpr (OUT_HALF)
            ((half_t*)Cv)[(size_t)rg * N + cg] = (half_t)acc[i][j][r];
          else
            ((float*)Cv)[(size_t)rg * N + cg] = acc[i][j][r];
        }
      }
}

// ---------------------------------------------------------------------------
// RoPE (interleaved pairs) applied in place to f16 q_r[B,S,16,64], k_r[B,S,64]
// ---------------------------------------------------------------------------
__global__ void rope_kernel(half_t* __restrict__ qr, half_t* __restrict__ kr,
                            int total, int S) {
  int idx = blockIdx.x * blockDim.x + threadIdx.x;
  if (idx >= total) return;
  int i   = idx & 31;   // pair index 0..31
  int tok = idx >> 5;   // b*S + s
  int s   = tok % S;

  float freq = (float)s * __powf(10000.0f, -(2.0f * (float)i) / 64.0f);
  float sn, cs;
  __sincosf(freq, &sn, &cs);

  {
    half_t* kp = kr + (size_t)tok * 64 + 2 * i;
    float re = (float)kp[0], im = (float)kp[1];
    kp[0] = (half_t)(re * cs - im * sn);
    kp[1] = (half_t)(re * sn + im * cs);
  }
#pragma unroll
  for (int hh = 0; hh < 16; ++hh) {
    half_t* qp = qr + ((size_t)tok * 16 + hh) * 64 + 2 * i;
    float re = (float)qp[0], im = (float)qp[1];
    qp[0] = (half_t)(re * cs - im * sn);
    qp[1] = (half_t)(re * sn + im * cs);
  }
}

// ---------------------------------------------------------------------------
// Flash-style causal attention. One wave per (b, head, 16-query tile).
//   q = [q_c | q_r] (d=192, 6 WMMA K-steps), kv tiles of 32 rows.
//   Scores via WMMA straight from global (k rows contiguous in d == B-frag
//   layout). Online softmax with shfl_xor reductions in 16-lane halves.
//   P goes through LDS (C-layout -> A-layout); V staged transposed in LDS.
// ---------------------------------------------------------------------------
__global__ __launch_bounds__(32) void mla_attn(
    const half_t* __restrict__ qc, const half_t* __restrict__ qr,
    const half_t* __restrict__ kc, const half_t* __restrict__ kr,
    const half_t* __restrict__ vb, half_t* __restrict__ attn, int S) {
  __shared__ alignas(16) half_t Pl[16 * 32];    // P tile, row-major [16][32]
  __shared__ alignas(16) half_t Vl[128 * 32];   // V tile transposed [dv][kv]

  const int lane = threadIdx.x & 31;
  const int hl   = lane >> 4;
  const int l16  = lane & 15;
  const int q0   = blockIdx.x * 16;
  const int h    = blockIdx.y;
  const int b    = blockIdx.z;
  const size_t tok0 = (size_t)b * S;

  // ---- load Q fragments (A layout), 6 K-steps over d=192 ----
  v16h aq[6];
  {
    const size_t t   = tok0 + q0 + l16;
    const half_t* qcp = qc + (t * 16 + h) * 128;
    const half_t* qrp = qr + (t * 16 + h) * 64;
#pragma unroll
    for (int st = 0; st < 4; ++st)
      aq[st] = pack16(*(const h8*)(qcp + 32 * st + 16 * hl),
                      *(const h8*)(qcp + 32 * st + 16 * hl + 8));
#pragma unroll
    for (int st = 0; st < 2; ++st)
      aq[4 + st] = pack16(*(const h8*)(qrp + 32 * st + 16 * hl),
                          *(const h8*)(qrp + 32 * st + 16 * hl + 8));
  }

  v8f O[8] = {};
  float mst[8], lst[8];
#pragma unroll
  for (int r = 0; r < 8; ++r) { mst[r] = -1e30f; lst[r] = 0.0f; }

  const float sc = 0.07216878f;  // 1/sqrt(192)

  for (int kv0 = 0; kv0 < q0 + 16; kv0 += 32) {
    // ---- scores: two 16x16 tiles, 6 WMMAs each ----
    v8f sv[2];
#pragma unroll
    for (int t = 0; t < 2; ++t) {
      v8f s = {};
      const size_t kt   = tok0 + kv0 + 16 * t + l16;
      const half_t* kcp = kc + (kt * 16 + h) * 128 + 16 * hl;
      const half_t* krp = kr + kt * 64 + 16 * hl;
#pragma unroll
      for (int st = 0; st < 4; ++st) {
        v16h bk = pack16(*(const h8*)(kcp + 32 * st),
                         *(const h8*)(kcp + 32 * st + 8));
        s = __builtin_amdgcn_wmma_f32_16x16x32_f16(false, aq[st], false, bk,
                                                   (short)0, s, false, false);
      }
#pragma unroll
      for (int st = 0; st < 2; ++st) {
        v16h bk = pack16(*(const h8*)(krp + 32 * st),
                         *(const h8*)(krp + 32 * st + 8));
        s = __builtin_amdgcn_wmma_f32_16x16x32_f16(false, aq[4 + st], false, bk,
                                                   (short)0, s, false, false);
      }
      sv[t] = s;
    }

    // ---- scale + causal mask + online softmax ----
    float p[2][8], tm[8];
#pragma unroll
    for (int r = 0; r < 8; ++r) {
      int row = q0 + r + 8 * hl;
#pragma unroll
      for (int t = 0; t < 2; ++t) {
        int col = kv0 + 16 * t + l16;
        p[t][r] = (col <= row) ? sv[t][r] * sc : -1e30f;
      }
      tm[r] = fmaxf(p[0][r], p[1][r]);
    }
#pragma unroll
    for (int off = 1; off <= 8; off <<= 1)
#pragma unroll
      for (int r = 0; r < 8; ++r)
        tm[r] = fmaxf(tm[r], __shfl_xor(tm[r], off, 32));

    float al[8], rs[8];
#pragma unroll
    for (int r = 0; r < 8; ++r) {
      float mn = fmaxf(mst[r], tm[r]);
      al[r]   = __expf(mst[r] - mn);
      mst[r]  = mn;
      p[0][r] = __expf(p[0][r] - mn);
      p[1][r] = __expf(p[1][r] - mn);
      rs[r]   = p[0][r] + p[1][r];
    }
#pragma unroll
    for (int off = 1; off <= 8; off <<= 1)
#pragma unroll
      for (int r = 0; r < 8; ++r)
        rs[r] += __shfl_xor(rs[r], off, 32);
#pragma unroll
    for (int r = 0; r < 8; ++r) lst[r] = lst[r] * al[r] + rs[r];
#pragma unroll
    for (int jt = 0; jt < 8; ++jt)
#pragma unroll
      for (int r = 0; r < 8; ++r) O[jt][r] *= al[r];

    // ---- P (C layout) -> LDS row-major ----
#pragma unroll
    for (int r = 0; r < 8; ++r) {
      Pl[(r + 8 * hl) * 32 + l16]      = (half_t)p[0][r];
      Pl[(r + 8 * hl) * 32 + 16 + l16] = (half_t)p[1][r];
    }
    // ---- V tile (32 kv rows x 128 dv) -> LDS transposed [dv][kv] ----
    for (int kk = 0; kk < 32; ++kk) {
      const h4 vv =
          *(const h4*)(vb + ((tok0 + kv0 + kk) * 16 + h) * 128 + lane * 4);
#pragma unroll
      for (int j = 0; j < 4; ++j) Vl[(lane * 4 + j) * 32 + kk] = vv[j];
    }
    __syncthreads();

    // ---- O += P @ V : one A-frag, 8 B-frags / WMMAs ----
    v16h pa;
    {
      const h8* pp = (const h8*)&Pl[l16 * 32];
      pa = pack16(pp[2 * hl], pp[2 * hl + 1]);
    }
#pragma unroll
    for (int jt = 0; jt < 8; ++jt) {
      const h8* pp = (const h8*)&Vl[(16 * jt + l16) * 32 + 16 * hl];
      v16h bv = pack16(pp[0], pp[1]);
      O[jt] = __builtin_amdgcn_wmma_f32_16x16x32_f16(false, pa, false, bv,
                                                     (short)0, O[jt], false,
                                                     false);
    }
    __syncthreads();
  }

  // ---- normalize and write attn[b, s, h, dv] as f16 ----
#pragma unroll
  for (int jt = 0; jt < 8; ++jt)
#pragma unroll
    for (int r = 0; r < 8; ++r) {
      int row = q0 + r + 8 * hl;
      int col = 16 * jt + l16;
      attn[((tok0 + row) * 16 + h) * 128 + col] = (half_t)(O[jt][r] / lst[r]);
    }
}

// ---------------------------------------------------------------------------
// Launch: 7 projection GEMMs -> RoPE -> attention -> output GEMM (f32 out)
// ---------------------------------------------------------------------------
extern "C" void kernel_launch(void* const* d_in, const int* in_sizes, int n_in,
                              void* d_out, int out_size, void* d_ws,
                              size_t ws_size, hipStream_t stream) {
  const float* x    = (const float*)d_in[0];
  const float* W_c  = (const float*)d_in[1];
  const float* W_cp = (const float*)d_in[2];
  const float* W_qc = (const float*)d_in[3];
  const float* W_qr = (const float*)d_in[4];
  const float* W_kc = (const float*)d_in[5];
  const float* W_kr = (const float*)d_in[6];
  const float* W_v  = (const float*)d_in[7];
  const float* W_o  = (const float*)d_in[8];

  const int S  = 2048;
  const int BS = in_sizes[0] / 2048;  // B*S tokens
  const int B  = BS / S;

  // f16 workspace layout
  half_t* c_  = (half_t*)d_ws;                    // [BS, 512]
  half_t* cp_ = c_  + (size_t)BS * 512;           // [BS, 1536]
  half_t* qc_ = cp_ + (size_t)BS * 1536;          // [BS, 16, 128]
  half_t* qr_ = qc_ + (size_t)BS * 2048;          // [BS, 16, 64]
  half_t* kc_ = qr_ + (size_t)BS * 1024;          // [BS, 16, 128]
  half_t* kr_ = kc_ + (size_t)BS * 2048;          // [BS, 64]
  half_t* v_  = kr_ + (size_t)BS * 64;            // [BS, 16, 128]
  half_t* at_ = v_  + (size_t)BS * 2048;          // [BS, 16, 128]

  dim3 blk(256);
  gemm_wmma<float,  true><<<dim3(512  / 128, BS / 64), blk, 0, stream>>>(x,   W_c,  c_,  BS, 512,  2048);
  gemm_wmma<float,  true><<<dim3(1536 / 128, BS / 64), blk, 0, stream>>>(x,   W_cp, cp_, BS, 1536, 2048);
  gemm_wmma<half_t, true><<<dim3(2048 / 128, BS / 64), blk, 0, stream>>>(cp_, W_qc, qc_, BS, 2048, 1536);
  gemm_wmma<half_t, true><<<dim3(1024 / 128, BS / 64), blk, 0, stream>>>(cp_, W_qr, qr_, BS, 1024, 1536);
  gemm_wmma<half_t, true><<<dim3(2048 / 128, BS / 64), blk, 0, stream>>>(c_,  W_kc, kc_, BS, 2048, 512);
  gemm_wmma<float,  true><<<dim3(1,          BS / 64), blk, 0, stream>>>(x,   W_kr, kr_, BS, 64,   2048);
  gemm_wmma<half_t, true><<<dim3(2048 / 128, BS / 64), blk, 0, stream>>>(c_,  W_v,  v_,  BS, 2048, 512);

  int total = BS * 32;
  rope_kernel<<<(total + 255) / 256, 256, 0, stream>>>(qr_, kr_, total, S);

  mla_attn<<<dim3(S / 16, 16, B), dim3(32), 0, stream>>>(qc_, qr_, kc_, kr_, v_, at_, S);

  gemm_wmma<half_t, false><<<dim3(2048 / 128, BS / 64), blk, 0, stream>>>(at_, W_o, (float*)d_out, BS, 2048, 2048);
}
